// RNN_38817914421676
// MI455X (gfx1250) — matrix operand order repaired and checked by reading, not compiled, not run on hardware
//
#include <hip/hip_runtime.h>

// ---------------------------------------------------------------------------
// IndRNN stack (2 layers) for MI455X / gfx1250.
//   B=32, T=1024, IN=H=512.
//   GEMMs via v_wmma_f32_16x16x32_f16 (f16 operands, f32 accumulate),
//   one wave per 16x64 output strip (4 WMMA tiles share one A fragment).
//   Recurrence: one thread per (b,h) scalar chain (16384 chains, wave32).
// ---------------------------------------------------------------------------

#define BB 32
#define TT 1024
#define HH 512            // IN == H == 512
#define MROWS (BB * TT)   // 32768 GEMM rows

typedef __attribute__((ext_vector_type(16))) _Float16 v16h;
typedef __attribute__((ext_vector_type(8)))  float    v8f;
typedef __attribute__((ext_vector_type(4)))  float    v4f;

// ---------------------------------------------------------------------------
// Convert f32 weight matrix [512x512] to f16, transposed (Wf[n][k]):
// WMMA B-fragment layout wants column N of W contiguous along K per lane).
// ---------------------------------------------------------------------------
__global__ void convert_w_f16(const float* __restrict__ W,
                              _Float16* __restrict__ Wf) {
    int i = blockIdx.x * blockDim.x + threadIdx.x;   // exact grid: 512*512
    Wf[(size_t)(i & (HH - 1)) * HH + (i >> 9)] = (_Float16)W[i];
}

// ---------------------------------------------------------------------------
// LayerNorm over rows of 512, applied `passes` times (1 or 2) with the same
// gamma/beta (reference applies LN0 twice), output converted to f16 for WMMA.
// One 256-thread block per row, 2 elements/thread.
// ---------------------------------------------------------------------------
__global__ void ln_rows_to_f16(const float* __restrict__ x,
                               const float* __restrict__ g,
                               const float* __restrict__ be,
                               _Float16* __restrict__ out,
                               int passes) {
    __shared__ float s_sum[8], s_sq[8], s_stats[2];
    const int row = blockIdx.x;
    const int tid = threadIdx.x;
    const float* xr = x + (size_t)row * HH;

    float x0 = xr[tid], x1 = xr[tid + 256];
    const float g0v = g[tid],  g1v = g[tid + 256];
    const float b0v = be[tid], b1v = be[tid + 256];

    for (int p = 0; p < passes; ++p) {
        float s = x0 + x1;
        float q = x0 * x0 + x1 * x1;
        #pragma unroll
        for (int off = 16; off > 0; off >>= 1) {
            s += __shfl_down(s, off, 32);
            q += __shfl_down(q, off, 32);
        }
        const int wave = tid >> 5;
        if ((tid & 31) == 0) { s_sum[wave] = s; s_sq[wave] = q; }
        __syncthreads();
        if (tid == 0) {
            float ts = 0.f, tq = 0.f;
            #pragma unroll
            for (int w = 0; w < 8; ++w) { ts += s_sum[w]; tq += s_sq[w]; }
            float m = ts * (1.0f / HH);
            float v = tq * (1.0f / HH) - m * m;
            s_stats[0] = m;
            s_stats[1] = rsqrtf(v + 1e-5f);
        }
        __syncthreads();
        const float m = s_stats[0], rs = s_stats[1];
        x0 = (x0 - m) * rs * g0v + b0v;
        x1 = (x1 - m) * rs * g1v + b1v;
        __syncthreads();   // protect s_* before next pass
    }
    out[(size_t)row * HH + tid]       = (_Float16)x0;
    out[(size_t)row * HH + tid + 256] = (_Float16)x1;
}

// ---------------------------------------------------------------------------
// GEMM: C[32768x512] = A[32768x512](f16) * W[512x512](f16) + bias, f32 out.
// One wave per 16(M) x 64(N) strip: per K-step (32), load A fragment once,
// issue 4 WMMAs against 4 B fragments (two 16B loads each from Wf row n = W column n).
//   A frag: lane<16 -> row M=lane, elems {K..K+7, K+16..K+23}; lanes>=16
//           same rows, K offsets +8 (two 16B contiguous loads).
//   B frag j: lane L -> W[k][n0+16j+(L&15)], k in same order as A frag.
//   C/D:    VGPR v, lane L -> (M = v + 8*(L>=16), N = 16j + (L&15)).
// ---------------------------------------------------------------------------
__global__ void wmma_gemm_bias(const _Float16* __restrict__ A,
                               const _Float16* __restrict__ Wf,
                               const float* __restrict__ bias,
                               float* __restrict__ C) {
    const int lane = threadIdx.x & 31;
    const int wave = threadIdx.x >> 5;
    const int strip = blockIdx.x * 8 + wave;  // 2048*8 = 16384 strips
    const int gm = strip >> 3;                // 2048 strips in M
    const int n0 = (strip & 7) * 64;          // 8 strips in N
    const int r  = lane & 15;
    const int hs = lane >> 4;

    const _Float16* aRow = A  + (size_t)(gm * 16 + r) * HH;
    const _Float16* bRow = Wf + (size_t)(n0 + r) * HH;

    v8f acc0 = {}, acc1 = {}, acc2 = {}, acc3 = {};
    #pragma unroll 2
    for (int k0 = 0; k0 < HH; k0 += 32) {
        v16h a;
        v4f* ap = (v4f*)&a;
        ap[0] = *(const v4f*)(aRow + k0 + hs * 8);
        ap[1] = *(const v4f*)(aRow + k0 + 16 + hs * 8);

        v16h b0, b1, b2, b3;
        const _Float16* bp = bRow + k0 + hs * 8;
        ((v4f*)&b0)[0] = *(const v4f*)(bp);               ((v4f*)&b0)[1] = *(const v4f*)(bp + 16);
        ((v4f*)&b1)[0] = *(const v4f*)(bp + 16 * HH);     ((v4f*)&b1)[1] = *(const v4f*)(bp + 16 * HH + 16);
        ((v4f*)&b2)[0] = *(const v4f*)(bp + 32 * HH);     ((v4f*)&b2)[1] = *(const v4f*)(bp + 32 * HH + 16);
        ((v4f*)&b3)[0] = *(const v4f*)(bp + 48 * HH);     ((v4f*)&b3)[1] = *(const v4f*)(bp + 48 * HH + 16);

        acc0 = __builtin_amdgcn_wmma_f32_16x16x32_f16(false, a, false, b0,
                                                      (short)0, acc0, false, false);
        acc1 = __builtin_amdgcn_wmma_f32_16x16x32_f16(false, a, false, b1,
                                                      (short)0, acc1, false, false);
        acc2 = __builtin_amdgcn_wmma_f32_16x16x32_f16(false, a, false, b2,
                                                      (short)0, acc2, false, false);
        acc3 = __builtin_amdgcn_wmma_f32_16x16x32_f16(false, a, false, b3,
                                                      (short)0, acc3, false, false);
    }

    float* cBase = C + (size_t)(gm * 16 + 8 * hs) * HH + n0 + r;
    #pragma unroll
    for (int j = 0; j < 4; ++j) {
        const v8f acc = (j == 0) ? acc0 : (j == 1) ? acc1 : (j == 2) ? acc2 : acc3;
        const float bv = bias[n0 + j * 16 + r];
        float* cp = cBase + j * 16;
        #pragma unroll
        for (int v = 0; v < 8; ++v)
            cp[(size_t)v * HH] = acc[v] + bv;
    }
}

// ---------------------------------------------------------------------------
// IndRNN recurrence: h_t = tanh(xwb[b][t][h] + u[h]*h_{t-1}).
// One thread per (b,h) chain. accumulate==0: y = h (layer 0 writes y0);
// accumulate==1: y += h (layer 1 adds y1 onto y0 already in d_out).
// Final h written to hT (tail of d_out).
// ---------------------------------------------------------------------------
__global__ void indrnn_scan(const float* __restrict__ xwb,
                            const float* __restrict__ u,
                            const float* __restrict__ h0,
                            float* __restrict__ y,
                            float* __restrict__ hT,
                            int accumulate) {
    const int idx = blockIdx.x * blockDim.x + threadIdx.x;  // exact: B*H
    const int b = idx >> 9;          // / 512
    const int h = idx & (HH - 1);
    const float uu = u[h];
    float hv = h0[idx];
    const float* xp = xwb + ((size_t)b << 19) + h;   // b*T*H
    float*       yp = y   + ((size_t)b << 19) + h;

    if (accumulate) {
        #pragma unroll 4
        for (int t = 0; t < TT; ++t) {
            hv = tanhf(xp[(size_t)t << 9] + uu * hv);
            yp[(size_t)t << 9] = yp[(size_t)t << 9] + hv;
        }
    } else {
        #pragma unroll 4
        for (int t = 0; t < TT; ++t) {
            hv = tanhf(xp[(size_t)t << 9] + uu * hv);
            yp[(size_t)t << 9] = hv;
        }
    }
    hT[idx] = hv;
}

// ---------------------------------------------------------------------------
extern "C" void kernel_launch(void* const* d_in, const int* in_sizes, int n_in,
                              void* d_out, int out_size, void* d_ws, size_t ws_size,
                              hipStream_t stream) {
    const float* x     = (const float*)d_in[0];
    const float* state = (const float*)d_in[1];   // [2, B, H]
    const float* W0    = (const float*)d_in[2];
    const float* u0    = (const float*)d_in[3];
    const float* b0    = (const float*)d_in[4];
    const float* g0    = (const float*)d_in[5];
    const float* be0   = (const float*)d_in[6];
    const float* W1    = (const float*)d_in[7];
    const float* u1    = (const float*)d_in[8];
    const float* b1    = (const float*)d_in[9];
    const float* g1    = (const float*)d_in[10];
    const float* be1   = (const float*)d_in[11];

    float* out = (float*)d_out;                    // [B,T,H] then [2,B,H]
    float* hT0 = out + (size_t)MROWS * HH;
    float* hT1 = hT0 + BB * HH;

    // Workspace layout (~97 MB total):
    char* ws = (char*)d_ws;
    _Float16* aF16 = (_Float16*)ws;                                // 32 MB
    float*    xwb  = (float*)(ws + (size_t)33554432);              // 64 MB
    _Float16* w0f  = (_Float16*)(ws + (size_t)33554432 + 67108864);// 0.5 MB
    _Float16* w1f  = w0f + (size_t)HH * HH;                        // 0.5 MB

    // Weight conversion to f16 (transposed to Wf[n][k], K contiguous per W column)
    convert_w_f16<<<(HH * HH) / 256, 256, 0, stream>>>(W0, w0f);
    convert_w_f16<<<(HH * HH) / 256, 256, 0, stream>>>(W1, w1f);

    // Layer 0: in0 = LN(LN(x)) -> f16
    ln_rows_to_f16<<<MROWS, 256, 0, stream>>>(x, g0, be0, aF16, 2);
    // xwb0 = in0 @ W0 + b0   (16384 strips / 8 waves per block)
    wmma_gemm_bias<<<2048, 256, 0, stream>>>(aF16, w0f, b0, xwb);
    // y0 scan -> d_out, final h0 -> tail
    indrnn_scan<<<(BB * HH) / 256, 256, 0, stream>>>(xwb, u0, state, out, hT0, 0);

    // Layer 1: in1 = LN(y0) -> f16
    ln_rows_to_f16<<<MROWS, 256, 0, stream>>>(out, g1, be1, aF16, 1);
    // xwb1 = in1 @ W1 + b1
    wmma_gemm_bias<<<2048, 256, 0, stream>>>(aF16, w1f, b1, xwb);
    // y1 scan, out += y1, final h1 -> tail
    indrnn_scan<<<(BB * HH) / 256, 256, 0, stream>>>(xwb, u1, state + BB * HH, out, hT1, 1);
}
